// MinkUNet_54193897341017
// MI455X (gfx1250) — compile-verified, matrix operand order checked by reading, not run on hardware
//
#include <hip/hip_runtime.h>

typedef __bf16 bf16;
typedef __attribute__((ext_vector_type(16))) __bf16 v16bf;
typedef __attribute__((ext_vector_type(8)))  __bf16 v8bf;
typedef __attribute__((ext_vector_type(8)))  float  v8f;
typedef __attribute__((ext_vector_type(4)))  int    v4i;

#if defined(__has_builtin)
# if __has_builtin(__builtin_amdgcn_global_load_async_to_lds_b128)
#  define HAVE_ASYNC_LDS 1
# else
#  define HAVE_ASYNC_LDS 0
# endif
#else
# define HAVE_ASYNC_LDS 0
#endif

__device__ __forceinline__ void wait_async_then_barrier()
{
#if HAVE_ASYNC_LDS
# if __has_builtin(__builtin_amdgcn_s_wait_asynccnt)
    __builtin_amdgcn_s_wait_asynccnt(0);
# else
    asm volatile("s_wait_asynccnt 0" ::: "memory");
# endif
#endif
    __syncthreads();
}

// ---------------------------------------------------------------------------
// 3x3x3 SAME conv + folded-BN + ReLU + mask as implicit GEMM on WMMA.
// Inputs live in zero-halo padded buffers [(D+2)^3, CIN] so the K-loop has no
// bounds checks. One block = one 16-voxel tile (16 consecutive w positions);
// COUT/16 waves per block, wave nt computes cout tile nt. All waves first
// cooperatively stage the 3x3x18 x CIN input slab into LDS (async global->LDS
// when available), then run the K-loop with A from LDS (ds_load_b128, padded
// stride CIN+8 to spread banks) and pre-packed B fragments from global.
//   A-fragment (ISA 16-bit A layout): lane L -> row r=L&15, K-half kb=(L>>4)*8,
//   klocal {kb..kb+7, kb+16..kb+23}. C/D: lane L, vgpr j -> (M=j+(L>>4)*8, N=L&15).
// ---------------------------------------------------------------------------
template<int CIN, int COUT, int D, bool OUTPAD>
__global__ __launch_bounds__(32 * (COUT / 16)) void conv3x3_bn_relu_wmma(
    const bf16* __restrict__ in,          // padded [(D+2)^3, CIN]
    const bf16* __restrict__ wpk,         // packed [COUT/16][27*CIN/32][32][16]
    const float* __restrict__ bnA, const float* __restrict__ bnB,
    const float* __restrict__ mask,       // [D^3] (unpadded)
    bf16* __restrict__ out, int outStride, int chanOff)
{
    constexpr int C32 = CIN / 32;
    constexpr int DP  = D + 2;
    constexpr int VS  = CIN + 8;          // LDS voxel stride (bank spread)
    constexpr int VEC = CIN / 8;          // v8bf chunks per voxel
    constexpr int NTHREADS = 32 * (COUT / 16);
    __shared__ bf16 smem[9 * 18 * VS];

    const int lane = threadIdx.x & 31;
    const int nt   = threadIdx.x >> 5;    // cout tile = wave id
    const int base = blockIdx.x * 16;
    const int d  = base / (D * D);
    const int hw = base % (D * D);
    const int h  = hw / D;
    const int w0 = hw % D;

    // ---- stage 3(d) x 3(h) x 18(w) x CIN slab into LDS ----
    for (int t = threadIdx.x; t < 9 * 18 * VEC; t += NTHREADS) {
        const int vec = t % VEC;
        const int vox = (t / VEC) % 18;
        const int seg = t / (18 * VEC);          // (kd+1)*3 + (kh+1)
        const int pd = d + seg / 3;              // padded coords
        const int ph = h + seg % 3;
        const int pw = w0 + vox;
        const size_t src = ((size_t)(pd * DP + ph) * DP + pw) * CIN + vec * 8;
        bf16* dst = &smem[(seg * 18 + vox) * VS + vec * 8];
#if HAVE_ASYNC_LDS
        __builtin_amdgcn_global_load_async_to_lds_b128(
            (v4i*)(in + src), (v4i*)dst, 0, 0);
#else
        *(v8bf*)dst = *(const v8bf*)(in + src);
#endif
    }
    wait_async_then_barrier();

    // ---- K-loop: 27 taps x C32 chunks of 32 bf16 ----
    const int r  = lane & 15;
    const int kb = (lane >> 4) * 8;
    v8f acc = {};
    const bf16* wp = wpk + ((size_t)nt * (27 * C32) * 32 + lane) * 16;

    int chunk = 0;
    for (int seg = 0; seg < 9; ++seg) {
        #pragma unroll
        for (int kw = 0; kw < 3; ++kw) {
            const bf16* arow = &smem[(seg * 18 + r + kw) * VS + kb];
            #pragma unroll
            for (int c = 0; c < C32; ++c, ++chunk) {
                union { v16bf v; v8bf q[2]; } A;
                A.q[0] = *(const v8bf*)(arow + c * 32);        // klocal kb..kb+7
                A.q[1] = *(const v8bf*)(arow + c * 32 + 16);   // klocal kb+16..+23
                const v16bf B = *(const v16bf*)(wp + (size_t)chunk * 512);
                acc = __builtin_amdgcn_wmma_f32_16x16x32_bf16(
                    false, A.v, false, B, (short)0, acc, false, false);
            }
        }
    }

    // ---- epilogue: BN + ReLU + mask, bf16 store (padded or flat) ----
    const int n = nt * 16 + (lane & 15);
    const float a = bnA[n], b = bnB[n];
    const int m0 = (lane >> 4) * 8;
    #pragma unroll
    for (int j = 0; j < 8; ++j) {
        const int wv = w0 + m0 + j;
        const int vlin = base + m0 + j;          // unpadded linear voxel
        float val = acc[j] * a + b;
        val = val > 0.f ? val : 0.f;
        val *= mask[vlin];
        size_t oi;
        if (OUTPAD)
            oi = ((size_t)((d + 1) * DP + (h + 1)) * DP + (wv + 1)) * outStride + chanOff + n;
        else
            oi = (size_t)vlin * outStride + chanOff + n;
        out[oi] = (bf16)val;
    }
}

// Pack f32 weights (k,k,k,Cin,Cout row-major) into WMMA B-fragment layout:
// dst[((nt*27*C32 + chunk)*32 + lane)*16 + j], chunk = tap*C32 + c32.
__global__ void pack_wmma_weights(const float* __restrict__ w, bf16* __restrict__ dst,
                                  int Cin, int Cout)
{
    const int idx = blockIdx.x * blockDim.x + threadIdx.x;
    const int C32 = Cin / 32;
    const int total = (Cout / 16) * 27 * C32 * 32 * 16;
    if (idx >= total) return;
    const int j     = idx & 15;
    const int lane  = (idx >> 4) & 31;
    const int chunk = (idx >> 9) % (27 * C32);
    const int nt    = (idx >> 9) / (27 * C32);
    const int kb = (lane >> 4) * 8;
    const int klocal = (j < 8) ? (kb + j) : (kb + 16 + (j - 8));
    const int tap = chunk / C32;
    const int cin = (chunk % C32) * 32 + klocal;
    const int n = nt * 16 + (lane & 15);
    dst[idx] = (bf16)w[((size_t)tap * Cin + cin) * Cout + n];
}

// ---------------------------------------------------------------------------
// Helper kernels (memory-bound layers; ~7% of FLOPs)
// ---------------------------------------------------------------------------
__global__ void bn_prep(const float* g, const float* b, const float* mean,
                        const float* var, float* A, float* B, int C)
{
    const int c = blockIdx.x * blockDim.x + threadIdx.x;
    if (c < C) {
        const float a = g[c] * rsqrtf(var[c] + 1e-5f);
        A[c] = a;
        B[c] = b[c] - mean[c] * a;
    }
}

__global__ void cvt_bf16(const float* __restrict__ s, bf16* __restrict__ d, int n)
{
    const int i = blockIdx.x * blockDim.x + threadIdx.x;
    if (i < n) d[i] = (bf16)s[i];
}

// Zero the halo of a padded [(D+2)^3, C] buffer (C multiple of 8).
__global__ void zero_halo(bf16* __restrict__ buf, int D, int C)
{
    const int DP = D + 2;
    const int vecs = C / 8;
    const int idx = blockIdx.x * blockDim.x + threadIdx.x;
    if (idx >= DP * DP * DP * vecs) return;
    const int vec = idx % vecs;
    const int v = idx / vecs;
    const int pd = v / (DP * DP);
    const int r = v % (DP * DP);
    const int ph = r / DP;
    const int pw = r % DP;
    if (pd == 0 || pd == DP - 1 || ph == 0 || ph == DP - 1 || pw == 0 || pw == DP - 1) {
        const v8bf z = {};
        *(v8bf*)(buf + (size_t)v * C + vec * 8) = z;
    }
}

__global__ void mask_input(const float* __restrict__ x, const int* __restrict__ m,
                           float* __restrict__ m1f, bf16* __restrict__ xm)
{
    const int v = blockIdx.x * blockDim.x + threadIdx.x;
    if (v >= 64 * 64 * 64) return;
    const float mv = m[v] ? 1.f : 0.f;
    m1f[v] = mv;
    #pragma unroll
    for (int c = 0; c < 4; ++c)
        xm[(size_t)v * 4 + c] = (bf16)(x[(size_t)v * 4 + c] * mv);
}

__global__ void pool_mask_i(const int* __restrict__ m, float* __restrict__ out, int DO)
{
    const int v = blockIdx.x * blockDim.x + threadIdx.x;
    if (v >= DO * DO * DO) return;
    const int DI = DO * 2;
    const int d = v / (DO * DO), hw = v % (DO * DO), h = hw / DO, w = hw % DO;
    int any = 0;
    for (int t = 0; t < 8; ++t) {
        const int di = 2 * d + (t >> 2), hi = 2 * h + ((t >> 1) & 1), wi = 2 * w + (t & 1);
        any |= m[(di * DI + hi) * DI + wi];
    }
    out[v] = any ? 1.f : 0.f;
}

__global__ void pool_mask_f(const float* __restrict__ m, float* __restrict__ out, int DO)
{
    const int v = blockIdx.x * blockDim.x + threadIdx.x;
    if (v >= DO * DO * DO) return;
    const int DI = DO * 2;
    const int d = v / (DO * DO), hw = v % (DO * DO), h = hw / DO, w = hw % DO;
    float any = 0.f;
    for (int t = 0; t < 8; ++t) {
        const int di = 2 * d + (t >> 2), hi = 2 * h + ((t >> 1) & 1), wi = 2 * w + (t & 1);
        any = fmaxf(any, m[(di * DI + hi) * DI + wi]);
    }
    out[v] = any > 0.f ? 1.f : 0.f;
}

// enc1: 3^3 conv 4->32 + BN + ReLU + mask (tiny K, memory bound -> VALU)
__global__ void enc1_conv(const bf16* __restrict__ in, const float* __restrict__ w,
                          const float* __restrict__ bnA, const float* __restrict__ bnB,
                          const float* __restrict__ mask, bf16* __restrict__ out)
{
    const int D = 64;
    const int idx = blockIdx.x * blockDim.x + threadIdx.x;
    const int n = idx & 31;
    const int v = idx >> 5;
    if (v >= D * D * D) return;
    const int d = v / (D * D);
    const int hw = v % (D * D);
    const int h = hw / D;
    const int wv = hw % D;
    float acc = 0.f;
    int tap = 0;
    for (int kd = -1; kd <= 1; ++kd)
        for (int kh = -1; kh <= 1; ++kh)
            for (int kw = -1; kw <= 1; ++kw, ++tap) {
                const int dn = d + kd, hn = h + kh, wn = wv + kw;
                if ((unsigned)dn < 64u && (unsigned)hn < 64u && (unsigned)wn < 64u) {
                    const bf16* ip = in + ((size_t)((dn * D + hn) * D + wn)) * 4;
                    const float* wp = w + (size_t)tap * 4 * 32 + n;
                    #pragma unroll
                    for (int c = 0; c < 4; ++c)
                        acc += (float)ip[c] * wp[c * 32];
                }
            }
    float val = acc * bnA[n] + bnB[n];
    val = val > 0.f ? val : 0.f;
    out[(size_t)v * 32 + n] = (bf16)(val * mask[v]);
}

// 2^3 stride-2 conv (pool path), * mask; writes PADDED [(DO+2)^3, C] output
template<int C, int DO>
__global__ void pool_conv_s2(const bf16* __restrict__ in, const bf16* __restrict__ w,
                             const float* __restrict__ mask, bf16* __restrict__ out)
{
    const int DI = DO * 2, DP = DO + 2;
    const int idx = blockIdx.x * blockDim.x + threadIdx.x;
    if (idx >= DO * DO * DO * C) return;
    const int n = idx % C;
    const int v = idx / C;
    const int d = v / (DO * DO);
    const int hw = v % (DO * DO);
    const int h = hw / DO;
    const int wv = hw % DO;
    float acc = 0.f;
    for (int t = 0; t < 8; ++t) {
        const int di = 2 * d + (t >> 2), hi = 2 * h + ((t >> 1) & 1), wi = 2 * wv + (t & 1);
        const bf16* ip = in + ((size_t)((di * DI + hi) * DI + wi)) * C;
        const bf16* wp = w + (size_t)t * C * C + n;
        for (int c = 0; c < C; ++c)
            acc += (float)ip[c] * (float)wp[c * C];
    }
    out[((size_t)((d + 1) * DP + (h + 1)) * DP + (wv + 1)) * C + n] = (bf16)(acc * mask[v]);
}

// k=2 s=2 transposed conv; writes into PADDED concat buffer at channel off 0
template<int CIN, int COUT, int DO>
__global__ void upsample_s2(const bf16* __restrict__ in, const bf16* __restrict__ w,
                            const float* __restrict__ mask, bf16* __restrict__ out,
                            int outStride)
{
    const int DI = DO / 2, DP = DO + 2;
    const int idx = blockIdx.x * blockDim.x + threadIdx.x;
    if (idx >= DO * DO * DO * COUT) return;
    const int n = idx % COUT;
    const int v = idx / COUT;
    const int d = v / (DO * DO);
    const int hw = v % (DO * DO);
    const int h = hw / DO;
    const int wv = hw % DO;
    const int t = ((d & 1) * 2 + (h & 1)) * 2 + (wv & 1);
    const bf16* ip = in + ((size_t)(((d >> 1) * DI + (h >> 1)) * DI + (wv >> 1))) * CIN;
    const bf16* wp = w + (size_t)t * CIN * COUT + n;
    float acc = 0.f;
    for (int c = 0; c < CIN; ++c)
        acc += (float)ip[c] * (float)wp[c * COUT];
    out[((size_t)((d + 1) * DP + (h + 1)) * DP + (wv + 1)) * outStride + n] =
        (bf16)(acc * mask[v]);
}

// copy flat [D^3, C] channels into PADDED [(D+2)^3, Ctot] at channel offset
__global__ void copy_ch_pad(const bf16* __restrict__ src, bf16* __restrict__ dst,
                            int C, int Ctot, int off, int D)
{
    const int idx = blockIdx.x * blockDim.x + threadIdx.x;
    if (idx >= D * D * D * C) return;
    const int c = idx % C, v = idx / C;
    const int DP = D + 2;
    const int d = v / (D * D), hw = v % (D * D), h = hw / D, w = hw % D;
    dst[((size_t)((d + 1) * DP + (h + 1)) * DP + (w + 1)) * Ctot + off + c] = src[idx];
}

// final 1x1x1 conv 32->3 + bias, * mask, f32 out
__global__ void final_conv(const bf16* __restrict__ in, const float* __restrict__ w,
                           const float* __restrict__ bias, const float* __restrict__ mask,
                           float* __restrict__ out)
{
    const int v = blockIdx.x * blockDim.x + threadIdx.x;
    if (v >= 64 * 64 * 64) return;
    float a0 = bias[0], a1 = bias[1], a2 = bias[2];
    const bf16* ip = in + (size_t)v * 32;
    #pragma unroll
    for (int c = 0; c < 32; ++c) {
        const float x = (float)ip[c];
        a0 += x * w[c * 3 + 0];
        a1 += x * w[c * 3 + 1];
        a2 += x * w[c * 3 + 2];
    }
    const float mv = mask[v];
    out[(size_t)v * 3 + 0] = a0 * mv;
    out[(size_t)v * 3 + 1] = a1 * mv;
    out[(size_t)v * 3 + 2] = a2 * mv;
}

// ---------------------------------------------------------------------------
// Host-side orchestration
// ---------------------------------------------------------------------------
static inline int ceil_div(int a, int b) { return (a + b - 1) / b; }

extern "C" void kernel_launch(void* const* d_in, const int* in_sizes, int n_in,
                              void* d_out, int out_size, void* d_ws, size_t ws_size,
                              hipStream_t stream)
{
    (void)in_sizes; (void)n_in; (void)out_size; (void)ws_size;

    const float* x       = (const float*)d_in[0];
    const int*   m1i     = (const int*)  d_in[1];
    const float* e1w     = (const float*)d_in[2];
    const float* e1g     = (const float*)d_in[3];
    const float* e1b     = (const float*)d_in[4];
    const float* e1m     = (const float*)d_in[5];
    const float* e1v     = (const float*)d_in[6];
    const float* pool1w  = (const float*)d_in[7];
    const float* e2w     = (const float*)d_in[8];
    const float* e2g     = (const float*)d_in[9];
    const float* e2b     = (const float*)d_in[10];
    const float* e2m     = (const float*)d_in[11];
    const float* e2v     = (const float*)d_in[12];
    const float* pool2w  = (const float*)d_in[13];
    const float* e3w     = (const float*)d_in[14];
    const float* e3g     = (const float*)d_in[15];
    const float* e3b     = (const float*)d_in[16];
    const float* e3m     = (const float*)d_in[17];
    const float* e3v     = (const float*)d_in[18];
    const float* d3w     = (const float*)d_in[19];
    const float* d3g     = (const float*)d_in[20];
    const float* d3b     = (const float*)d_in[21];
    const float* d3m     = (const float*)d_in[22];
    const float* d3v     = (const float*)d_in[23];
    const float* up2w    = (const float*)d_in[24];
    const float* d2w     = (const float*)d_in[25];
    const float* d2g     = (const float*)d_in[26];
    const float* d2b     = (const float*)d_in[27];
    const float* d2m     = (const float*)d_in[28];
    const float* d2v     = (const float*)d_in[29];
    const float* up1w    = (const float*)d_in[30];
    const float* d1w     = (const float*)d_in[31];
    const float* d1g     = (const float*)d_in[32];
    const float* d1b     = (const float*)d_in[33];
    const float* d1m     = (const float*)d_in[34];
    const float* d1v     = (const float*)d_in[35];
    const float* finw    = (const float*)d_in[36];
    const float* finb    = (const float*)d_in[37];

    const int V1 = 64 * 64 * 64, V2 = 32 * 32 * 32, V3 = 16 * 16 * 16;
    const int P1 = 66 * 66 * 66, P2 = 34 * 34 * 34, P3 = 18 * 18 * 18;

    char* p = (char*)d_ws;
    auto alloc = [&](size_t bytes) -> char* {
        char* r = p;
        p += (bytes + 255) & ~(size_t)255;
        return r;
    };

    bf16*  xm    = (bf16*) alloc((size_t)V1 * 4 * 2);
    float* m1f   = (float*)alloc((size_t)V1 * 4);
    float* m2    = (float*)alloc((size_t)V2 * 4);
    float* m3    = (float*)alloc((size_t)V3 * 4);
    bf16*  e1    = (bf16*) alloc((size_t)V1 * 32 * 2);        // flat
    bf16*  p1    = (bf16*) alloc((size_t)P2 * 32 * 2);        // padded (D=32)
    bf16*  e2    = (bf16*) alloc((size_t)V2 * 64 * 2);        // flat
    bf16*  p2    = (bf16*) alloc((size_t)P3 * 64 * 2);        // padded (D=16)
    bf16*  h3    = (bf16*) alloc((size_t)P3 * 128 * 2);       // padded (D=16)
    bf16*  hd3   = (bf16*) alloc((size_t)V3 * 64 * 2);        // flat
    bf16*  cat2  = (bf16*) alloc((size_t)P2 * 128 * 2);       // padded (D=32)
    bf16*  hd2   = (bf16*) alloc((size_t)V2 * 64 * 2);        // flat
    bf16*  cat1  = (bf16*) alloc((size_t)P1 * 64 * 2);        // padded (D=64)
    bf16*  hd1   = (bf16*) alloc((size_t)V1 * 32 * 2);        // flat
    bf16*  pk_e2 = (bf16*) alloc((size_t)27 * 32 * 64 * 2);
    bf16*  pk_e3 = (bf16*) alloc((size_t)27 * 64 * 128 * 2);
    bf16*  pk_d3 = (bf16*) alloc((size_t)27 * 128 * 64 * 2);
    bf16*  pk_d2 = (bf16*) alloc((size_t)27 * 128 * 64 * 2);
    bf16*  pk_d1 = (bf16*) alloc((size_t)27 * 64 * 32 * 2);
    bf16*  bw_p1 = (bf16*) alloc((size_t)8 * 32 * 32 * 2);
    bf16*  bw_p2 = (bf16*) alloc((size_t)8 * 64 * 64 * 2);
    bf16*  bw_u2 = (bf16*) alloc((size_t)8 * 64 * 64 * 2);
    bf16*  bw_u1 = (bf16*) alloc((size_t)8 * 64 * 32 * 2);
    float* bnA1  = (float*)alloc(32 * 4);  float* bnB1 = (float*)alloc(32 * 4);
    float* bnA2  = (float*)alloc(64 * 4);  float* bnB2 = (float*)alloc(64 * 4);
    float* bnA3  = (float*)alloc(128 * 4); float* bnB3 = (float*)alloc(128 * 4);
    float* bnAd3 = (float*)alloc(64 * 4);  float* bnBd3 = (float*)alloc(64 * 4);
    float* bnAd2 = (float*)alloc(64 * 4);  float* bnBd2 = (float*)alloc(64 * 4);
    float* bnAd1 = (float*)alloc(32 * 4);  float* bnBd1 = (float*)alloc(32 * 4);

    // --- parameter prep ---
    bn_prep<<<1, 128, 0, stream>>>(e1g, e1b, e1m, e1v, bnA1, bnB1, 32);
    bn_prep<<<1, 128, 0, stream>>>(e2g, e2b, e2m, e2v, bnA2, bnB2, 64);
    bn_prep<<<1, 128, 0, stream>>>(e3g, e3b, e3m, e3v, bnA3, bnB3, 128);
    bn_prep<<<1, 128, 0, stream>>>(d3g, d3b, d3m, d3v, bnAd3, bnBd3, 64);
    bn_prep<<<1, 128, 0, stream>>>(d2g, d2b, d2m, d2v, bnAd2, bnBd2, 64);
    bn_prep<<<1, 128, 0, stream>>>(d1g, d1b, d1m, d1v, bnAd1, bnBd1, 32);

    cvt_bf16<<<ceil_div(8 * 32 * 32, 256), 256, 0, stream>>>(pool1w, bw_p1, 8 * 32 * 32);
    cvt_bf16<<<ceil_div(8 * 64 * 64, 256), 256, 0, stream>>>(pool2w, bw_p2, 8 * 64 * 64);
    cvt_bf16<<<ceil_div(8 * 64 * 64, 256), 256, 0, stream>>>(up2w, bw_u2, 8 * 64 * 64);
    cvt_bf16<<<ceil_div(8 * 64 * 32, 256), 256, 0, stream>>>(up1w, bw_u1, 8 * 64 * 32);

    pack_wmma_weights<<<ceil_div(27 * 32 * 64, 256), 256, 0, stream>>>(e2w, pk_e2, 32, 64);
    pack_wmma_weights<<<ceil_div(27 * 64 * 128, 256), 256, 0, stream>>>(e3w, pk_e3, 64, 128);
    pack_wmma_weights<<<ceil_div(27 * 128 * 64, 256), 256, 0, stream>>>(d3w, pk_d3, 128, 64);
    pack_wmma_weights<<<ceil_div(27 * 128 * 64, 256), 256, 0, stream>>>(d2w, pk_d2, 128, 64);
    pack_wmma_weights<<<ceil_div(27 * 64 * 32, 256), 256, 0, stream>>>(d1w, pk_d1, 64, 32);

    // --- halo zeroing for padded conv inputs ---
    zero_halo<<<ceil_div(P2 * 4, 256), 256, 0, stream>>>(p1, 32, 32);
    zero_halo<<<ceil_div(P3 * 8, 256), 256, 0, stream>>>(p2, 16, 64);
    zero_halo<<<ceil_div(P3 * 16, 256), 256, 0, stream>>>(h3, 16, 128);
    zero_halo<<<ceil_div(P2 * 16, 256), 256, 0, stream>>>(cat2, 32, 128);
    zero_halo<<<ceil_div(P1 * 8, 256), 256, 0, stream>>>(cat1, 64, 64);

    // --- masks + masked input ---
    mask_input<<<ceil_div(V1, 256), 256, 0, stream>>>(x, m1i, m1f, xm);
    pool_mask_i<<<ceil_div(V2, 256), 256, 0, stream>>>(m1i, m2, 32);
    pool_mask_f<<<ceil_div(V3, 256), 256, 0, stream>>>(m2, m3, 16);

    // --- encoder ---
    enc1_conv<<<ceil_div(V1 * 32, 256), 256, 0, stream>>>(xm, e1w, bnA1, bnB1, m1f, e1);
    pool_conv_s2<32, 32><<<ceil_div(V2 * 32, 256), 256, 0, stream>>>(e1, bw_p1, m2, p1);
    conv3x3_bn_relu_wmma<32, 64, 32, false><<<V2 / 16, 128, 0, stream>>>(
        p1, pk_e2, bnA2, bnB2, m2, e2, 64, 0);
    pool_conv_s2<64, 16><<<ceil_div(V3 * 64, 256), 256, 0, stream>>>(e2, bw_p2, m3, p2);
    conv3x3_bn_relu_wmma<64, 128, 16, true><<<V3 / 16, 256, 0, stream>>>(
        p2, pk_e3, bnA3, bnB3, m3, h3, 128, 0);

    // --- bottleneck decode ---
    conv3x3_bn_relu_wmma<128, 64, 16, false><<<V3 / 16, 128, 0, stream>>>(
        h3, pk_d3, bnAd3, bnBd3, m3, hd3, 64, 0);

    // --- up2 + dec2 ---
    upsample_s2<64, 64, 32><<<ceil_div(V2 * 64, 256), 256, 0, stream>>>(hd3, bw_u2, m2, cat2, 128);
    copy_ch_pad<<<ceil_div(V2 * 64, 256), 256, 0, stream>>>(e2, cat2, 64, 128, 64, 32);
    conv3x3_bn_relu_wmma<128, 64, 32, false><<<V2 / 16, 128, 0, stream>>>(
        cat2, pk_d2, bnAd2, bnBd2, m2, hd2, 64, 0);

    // --- up1 + dec1 ---
    upsample_s2<64, 32, 64><<<ceil_div(V1 * 32, 256), 256, 0, stream>>>(hd2, bw_u1, m1f, cat1, 64);
    copy_ch_pad<<<ceil_div(V1 * 32, 256), 256, 0, stream>>>(e1, cat1, 32, 64, 32, 64);
    conv3x3_bn_relu_wmma<64, 32, 64, false><<<V1 / 16, 64, 0, stream>>>(
        cat1, pk_d1, bnAd1, bnBd1, m1f, hd1, 32, 0);

    // --- final 1x1x1 + bias + mask (f32 out) ---
    final_conv<<<ceil_div(V1, 256), 256, 0, stream>>>(hd1, finw, finb, m1f, (float*)d_out);
}
